// DGCNN_90228672954728
// MI455X (gfx1250) — compile-verified, matrix operand order checked by reading, not run on hardware
//
#include <hip/hip_runtime.h>
#include <hip/hip_bf16.h>
#include <stdint.h>

typedef __attribute__((ext_vector_type(16))) _Float16 v16h;
typedef __attribute__((ext_vector_type(8)))  float    v8f;
typedef __attribute__((ext_vector_type(4)))  unsigned v4u;
typedef __attribute__((ext_vector_type(8)))  int      v8i;
typedef __attribute__((ext_vector_type(4)))  int      v4i;

#define EPSBN  1e-5f
#define SLOPE  0.2f
#define BATCH  8
#define NPTS   2048
#define KNN    20

union AFrag { v16h v; _Float16 e[16]; };
union CFrag { v8f  v; float    e[8];  };

// K index inside a 16x32 f16 WMMA fragment for (lane, half-slot h)
__device__ __forceinline__ int kmap(int lane, int h) {
  return ((h & 8) << 1) + ((lane & 16) >> 1) + (h & 7);
}

// order-preserving f32 <-> u32 encoding (for atomicMax/Min on floats)
__device__ __forceinline__ unsigned enc_f32(float f) {
  unsigned u = __float_as_uint(f);
  return (u & 0x80000000u) ? ~u : (u | 0x80000000u);
}
__device__ __forceinline__ float dec_f32(unsigned e) {
  unsigned u = (e & 0x80000000u) ? (e ^ 0x80000000u) : ~e;
  return __uint_as_float(u);
}

// -----------------------------------------------------------------------------
// TDM: 2-D f16 tile load (rows x width halfs) global -> LDS.
// D# per CDNA5 ISA: group0 = {count=1 | lds_addr |
// global_addr | type=2}; group1 = {data_size=1 (2B), tensor_dim0/1, tile_dim0/1,
// tensor_dim0_stride}; groups 2/3 zero (<=2-D). Completion via TENSORcnt.
// clang-23 lane: 6-arg builtin (v4u, v8i, v4i, v4i, v8i, i32 cpol).
// -----------------------------------------------------------------------------
__device__ __forceinline__ void tdm_load_f16_2d(unsigned lds_byte_off,
                                                const _Float16* gptr,
                                                unsigned width, unsigned rows,
                                                unsigned stride, unsigned tensor_rows) {
  unsigned long long ga = (unsigned long long)(uintptr_t)gptr;
  v4u g0;
  g0[0] = 1u;                                               // count=1, user descriptor
  g0[1] = lds_byte_off;                                     // lds_addr
  g0[2] = (unsigned)(ga & 0xFFFFFFFFu);                     // global_addr[31:0]
  g0[3] = (unsigned)((ga >> 32) & 0x01FFFFFFu) | (2u << 30);// global_addr[56:32], type=2
  v8i g1;
  g1[0] = (int)(1u << 16);                                  // data_size=1 -> 2 bytes
  g1[1] = (int)((width & 0xFFFFu) << 16);                   // tensor_dim0[15:0]
  g1[2] = (int)(((width >> 16) & 0xFFFFu) |
                ((tensor_rows & 0xFFFFu) << 16));           // dim0[31:16] | dim1[15:0]
  g1[3] = (int)(((tensor_rows >> 16) & 0xFFFFu) |
                ((width & 0xFFFFu) << 16));                 // dim1[31:16] | tile_dim0
  g1[4] = (int)(rows & 0xFFFFu);                            // tile_dim1 (tile_dim2=0)
  g1[5] = (int)stride;                                      // tensor_dim0_stride[31:0]
  g1[6] = 0;
  g1[7] = 0;
  v4i z4 = {0, 0, 0, 0};
  v8i z8 = {0, 0, 0, 0, 0, 0, 0, 0};
  __builtin_amdgcn_tensor_load_to_lds(g0, g1, z4, z4, z8, 0);
}

// -----------------------------------------------------------------------------
// Prep: x (B,N,C) f32 -> xh (B,N,Cpad) f16 (zero-padded) + per-point |x|^2.
// -----------------------------------------------------------------------------
__global__ void prep_x_kernel(const float* __restrict__ x, int C, int Cpad,
                              _Float16* __restrict__ xh, float* __restrict__ sq) {
  int row = blockIdx.x * blockDim.x + threadIdx.x;
  if (row >= BATCH * NPTS) return;
  float s = 0.f;
  for (int c = 0; c < Cpad; ++c) {
    float v = (c < C) ? x[(size_t)row * C + c] : 0.f;
    s += v * v;
    xh[(size_t)row * Cpad + c] = (_Float16)v;
  }
  sq[row] = s;
}

// W (O,twoC) f32 -> Wh (O,twoCpad) f16 zero-padded.
__global__ void prep_w_kernel(const float* __restrict__ W, int O, int twoC, int twoCpad,
                              _Float16* __restrict__ Wh) {
  int i = blockIdx.x * blockDim.x + threadIdx.x;
  if (i >= O * twoCpad) return;
  int o = i / twoCpad, c = i % twoCpad;
  float v = (c < twoC) ? W[(size_t)o * twoC + c] : 0.f;
  Wh[i] = (_Float16)v;
}

// -----------------------------------------------------------------------------
// Kernel 1: neg_dist = 2*x.x^T - |xi|^2 - |xj|^2, one batch, WMMA f16.
// Operand tiles pulled into LDS by the Tensor Data Mover (one 16-row slice per
// wave), completed with s_wait_tensorcnt; inner loop = ds_load_b128 + v_wmma.
// Dynamic LDS layout: aT @0 (16 x Cpad f16), bT @32*Cpad (128 x Cpad f16).
// -----------------------------------------------------------------------------
__global__ void dist_wmma_kernel(const _Float16* __restrict__ xh,
                                 const float* __restrict__ sq,
                                 int b, int Cpad, float* __restrict__ dist) {
  extern __shared__ _Float16 dynsmem[];
  _Float16* aT = dynsmem;                // 16 x Cpad
  _Float16* bT = dynsmem + 16 * Cpad;    // 128 x Cpad

  const int tid  = threadIdx.x;
  const int lane = tid & 31;
  const int wave = tid >> 5;
  const int row0 = blockIdx.x * 16;
  const int colbase = blockIdx.y * 128;
  const _Float16* xb = xh + (size_t)b * NPTS * Cpad;

  // TDM: each wave DMAs its own 16-row B slice; wave 0 also brings the A tile.
  const unsigned aT_bytes = 32u * (unsigned)Cpad;          // 16*Cpad*2
  tdm_load_f16_2d(aT_bytes + (unsigned)wave * aT_bytes,
                  xb + (size_t)(colbase + wave * 16) * Cpad,
                  (unsigned)Cpad, 16u, (unsigned)Cpad, (unsigned)NPTS);
  if (wave == 0)
    tdm_load_f16_2d(0u, xb + (size_t)row0 * Cpad,
                    (unsigned)Cpad, 16u, (unsigned)Cpad, (unsigned)NPTS);
  __builtin_amdgcn_s_wait_tensorcnt(0);
  __syncthreads();

  const int n = lane & 15;
  v8f acc = {};
  const int ksteps = Cpad >> 5;
  for (int kc = 0; kc < ksteps; ++kc) {
    AFrag a, bb;
#pragma unroll
    for (int h = 0; h < 16; ++h) {
      int k = (kc << 5) + kmap(lane, h);
      a.e[h]  = aT[(lane & 15) * Cpad + k];            // A: M = lane%16
      bb.e[h] = bT[((wave << 4) + n) * Cpad + k];      // B(k,n) = x[col][k]
    }
    acc = __builtin_amdgcn_wmma_f32_16x16x32_f16(false, a.v, false, bb.v,
                                                 (short)0, acc, false, false);
  }

  CFrag cf; cf.v = acc;
  const int col  = colbase + (wave << 4) + n;
  const int mofs = (lane >> 4) << 3;                   // D: M = r (+8 hi lanes)
  const float sqn = sq[(size_t)b * NPTS + col];
#pragma unroll
  for (int r = 0; r < 8; ++r) {
    int m = r + mofs;
    float nd = 2.f * cf.e[r] - sq[(size_t)b * NPTS + row0 + m] - sqn;
    dist[(size_t)(row0 + m) * NPTS + col] = nd;
  }
}

// -----------------------------------------------------------------------------
// Kernel 2: top-20 per 2048-wide row; one wave32/row, 64 vals/lane in regs,
// 20 argmax passes w/ shfl_xor butterfly; ties -> lowest index (jax.lax.top_k).
// -----------------------------------------------------------------------------
__global__ void topk_kernel(const float* __restrict__ dist, int b, int* __restrict__ idx) {
  const int tid = threadIdx.x, lane = tid & 31, wave = tid >> 5;
  const int row = blockIdx.x * 8 + wave;
  const float* d = dist + (size_t)row * NPTS;
  float v[64];
#pragma unroll
  for (int j = 0; j < 64; ++j) v[j] = d[lane + 32 * j];
  int* out = idx + ((size_t)b * NPTS + row) * KNN;
  for (int p = 0; p < KNN; ++p) {
    float bv = -3.4e38f; int bc = NPTS;
#pragma unroll
    for (int j = 0; j < 64; ++j) {
      if (v[j] > bv) { bv = v[j]; bc = lane + 32 * j; }
    }
#pragma unroll
    for (int s = 16; s > 0; s >>= 1) {
      float ov = __shfl_xor(bv, s, 32);
      int   oc = __shfl_xor(bc, s, 32);
      if (ov > bv || (ov == bv && oc < bc)) { bv = ov; bc = oc; }
    }
    if (lane == 0) out[p] = bc;
#pragma unroll
    for (int j = 0; j < 64; ++j)
      if (lane + 32 * j == bc) v[j] = -3.4e38f;
  }
}

// -----------------------------------------------------------------------------
// Kernel 3: h = edge(x) @ W^T via WMMA. Edge rows gathered to LDS (f16);
// W tile (pre-converted f16) TDM-loaded per wave. Epilogue: hierarchical
// channel sum/sumsq atomics + per-(b,n,o) order-encoded max/min atomics.
// Dynamic LDS: aT @0 (16 x twoCpad), bW @32*twoCpad (128 x twoCpad).
// -----------------------------------------------------------------------------
__global__ void edgeconv_wmma_kernel(const float* __restrict__ xin, const int* __restrict__ idx,
                                     const _Float16* __restrict__ Wh,
                                     int Cin, int O, int twoCpad,
                                     float* __restrict__ sum, float* __restrict__ sumsq,
                                     unsigned* __restrict__ maxe, unsigned* __restrict__ mine) {
  extern __shared__ _Float16 dynsmem[];
  _Float16* aT = dynsmem;                      // 16 x twoCpad (gathered)
  _Float16* bW = dynsmem + 16 * twoCpad;       // 128 x twoCpad (TDM)

  const int tid = threadIdx.x, lane = tid & 31, wave = tid >> 5;
  const int m0 = blockIdx.x * 16;
  const int twoC = 2 * Cin;
  const int col0 = blockIdx.y * 128 + wave * 16;
  const bool active = col0 < O;
  const unsigned aT_bytes = 32u * (unsigned)twoCpad;

  if (active)   // wave-uniform; one TDM issue per wave for its 16 W rows
    tdm_load_f16_2d(aT_bytes + (unsigned)wave * aT_bytes,
                    Wh + (size_t)col0 * twoCpad,
                    (unsigned)twoCpad, 16u, (unsigned)twoCpad, (unsigned)O);

  for (int i = tid; i < 16 * twoCpad; i += blockDim.x) {
    int mr = i / twoCpad, cc = i % twoCpad;
    int m  = m0 + mr;
    int bb = m / (NPTS * KNN);
    int rem = m % (NPTS * KNN);
    int np = rem / KNN;
    int kk = rem % KNN;
    float v = 0.f;
    if (cc < Cin) {
      int nb = idx[((size_t)bb * NPTS + np) * KNN + kk];
      v = xin[((size_t)bb * NPTS + nb) * Cin + cc] - xin[((size_t)bb * NPTS + np) * Cin + cc];
    } else if (cc < twoC) {
      v = xin[((size_t)bb * NPTS + np) * Cin + (cc - Cin)];
    }
    aT[mr * twoCpad + cc] = (_Float16)v;
  }
  if (active) __builtin_amdgcn_s_wait_tensorcnt(0);
  __syncthreads();
  if (!active) return;

  const int n = lane & 15;
  v8f acc = {};
  const int ksteps = twoCpad >> 5;
  for (int kc = 0; kc < ksteps; ++kc) {
    AFrag a, bf;
#pragma unroll
    for (int h = 0; h < 16; ++h) {
      int k = (kc << 5) + kmap(lane, h);
      a.e[h]  = aT[(lane & 15) * twoCpad + k];
      bf.e[h] = bW[((wave << 4) + n) * twoCpad + k];   // B(k,n) = W[col0+n][k]
    }
    acc = __builtin_amdgcn_wmma_f32_16x16x32_f16(false, a.v, false, bf.v,
                                                 (short)0, acc, false, false);
  }

  CFrag cf; cf.v = acc;
  const int o = col0 + n;
  float s1 = 0.f, s2 = 0.f;
#pragma unroll
  for (int r = 0; r < 8; ++r) { float h = cf.e[r]; s1 += h; s2 += h * h; }
  s1 += __shfl_xor(s1, 16, 32);
  s2 += __shfl_xor(s2, 16, 32);
  if (lane < 16) { atomicAdd(&sum[o], s1); atomicAdd(&sumsq[o], s2); }

  const int mofs = (lane >> 4) << 3;
#pragma unroll
  for (int r = 0; r < 8; ++r) {
    int m = m0 + mofs + r;
    int bb = m / (NPTS * KNN);
    int rem = m % (NPTS * KNN);
    int np = rem / KNN;
    unsigned e = enc_f32(cf.e[r]);
    size_t off = ((size_t)bb * NPTS + np) * O + o;
    atomicMax(&maxe[off], e);
    atomicMin(&mine[off], e);
  }
}

// -----------------------------------------------------------------------------
// Kernel 4: BN (training stats) + LeakyReLU on max_k h (min_k if gamma<0).
// -----------------------------------------------------------------------------
__global__ void finalize_kernel(const unsigned* __restrict__ maxe, const unsigned* __restrict__ mine,
                                const float* __restrict__ sum, const float* __restrict__ sumsq,
                                const float* __restrict__ gamma, const float* __restrict__ beta,
                                int O, float invcnt,
                                float* __restrict__ out, int col_off,
                                float* __restrict__ xnext) {
  int i = blockIdx.x * blockDim.x + threadIdx.x;
  int total = BATCH * NPTS * O;
  if (i >= total) return;
  int o  = i % O;
  int pn = i / O;
  float mean = sum[o] * invcnt;
  float var  = sumsq[o] * invcnt - mean * mean;
  float g = gamma[o];
  float M = (g >= 0.f) ? dec_f32(maxe[i]) : dec_f32(mine[i]);
  float y = (M - mean) * rsqrtf(var + EPSBN) * g + beta[o];
  y = (y >= 0.f) ? y : SLOPE * y;
  out[(size_t)pn * 512 + col_off + o] = y;
  if (xnext) xnext[(size_t)pn * O + o] = y;
}

// -----------------------------------------------------------------------------
extern "C" void kernel_launch(void* const* d_in, const int* in_sizes, int n_in,
                              void* d_out, int out_size, void* d_ws, size_t ws_size,
                              hipStream_t stream) {
  (void)in_sizes; (void)n_in; (void)out_size; (void)ws_size;
  const float* x0 = (const float*)d_in[0];
  const float* W[4] = {(const float*)d_in[1], (const float*)d_in[4],
                       (const float*)d_in[7], (const float*)d_in[10]};
  const float* G[4] = {(const float*)d_in[2], (const float*)d_in[5],
                       (const float*)d_in[8], (const float*)d_in[11]};
  const float* Bt[4] = {(const float*)d_in[3], (const float*)d_in[6],
                        (const float*)d_in[9], (const float*)d_in[12]};
  // d_in[13] is k; known constant 20

  // workspace layout (~73 MB)
  char* ws = (char*)d_ws;
  float*     dist  = (float*)ws;     ws += (size_t)NPTS * NPTS * sizeof(float);
  int*       idx   = (int*)ws;       ws += (size_t)BATCH * NPTS * KNN * sizeof(int);
  float*     x1    = (float*)ws;     ws += (size_t)BATCH * NPTS * 64  * sizeof(float);
  float*     x2    = (float*)ws;     ws += (size_t)BATCH * NPTS * 64  * sizeof(float);
  float*     x3    = (float*)ws;     ws += (size_t)BATCH * NPTS * 128 * sizeof(float);
  unsigned*  maxe  = (unsigned*)ws;  ws += (size_t)BATCH * NPTS * 256 * sizeof(unsigned);
  unsigned*  mine  = (unsigned*)ws;  ws += (size_t)BATCH * NPTS * 256 * sizeof(unsigned);
  _Float16*  xh    = (_Float16*)ws;  ws += (size_t)BATCH * NPTS * 128 * sizeof(_Float16);
  _Float16*  Wh    = (_Float16*)ws;  ws += (size_t)256 * 256 * sizeof(_Float16);
  float*     sq    = (float*)ws;     ws += (size_t)BATCH * NPTS * sizeof(float);
  float*     sum   = (float*)ws;     ws += 256 * sizeof(float);
  float*     sumsq = (float*)ws;

  const int Cin[4]  = {3, 64, 64, 128};
  const int Och[4]  = {64, 64, 128, 256};
  const int coff[4] = {0, 64, 128, 256};
  float* xn[4] = {x1, x2, x3, nullptr};
  const float* xcur = x0;
  float* outp = (float*)d_out;

  for (int blk = 0; blk < 4; ++blk) {
    const int C = Cin[blk], O = Och[blk];
    const int Cpad    = (C + 31) & ~31;
    const int twoC    = 2 * C;
    const int twoCpad = (twoC + 31) & ~31;

    prep_x_kernel<<<(BATCH * NPTS + 255) / 256, 256, 0, stream>>>(xcur, C, Cpad, xh, sq);
    prep_w_kernel<<<(O * twoCpad + 255) / 256, 256, 0, stream>>>(W[blk], O, twoC, twoCpad, Wh);

    const size_t dist_lds = (size_t)144 * Cpad * sizeof(_Float16);   // (16+128) rows
    for (int b = 0; b < BATCH; ++b) {
      dim3 gd(NPTS / 16, NPTS / 128);
      dist_wmma_kernel<<<gd, 256, dist_lds, stream>>>(xh, sq, b, Cpad, dist);
      topk_kernel<<<NPTS / 8, 256, 0, stream>>>(dist, b, idx);
    }

    (void)hipMemsetAsync(sum,   0,    O * sizeof(float), stream);
    (void)hipMemsetAsync(sumsq, 0,    O * sizeof(float), stream);
    (void)hipMemsetAsync(maxe,  0x00, (size_t)BATCH * NPTS * O * sizeof(unsigned), stream);
    (void)hipMemsetAsync(mine,  0xFF, (size_t)BATCH * NPTS * O * sizeof(unsigned), stream);

    const size_t conv_lds = (size_t)144 * twoCpad * sizeof(_Float16);
    dim3 gc(BATCH * NPTS * KNN / 16, (O + 127) / 128);
    edgeconv_wmma_kernel<<<gc, 256, conv_lds, stream>>>(xcur, idx, Wh, C, O, twoCpad,
                                                        sum, sumsq, maxe, mine);

    const float invcnt = 1.f / (float)(BATCH * NPTS * KNN);
    const int total = BATCH * NPTS * O;
    finalize_kernel<<<(total + 255) / 256, 256, 0, stream>>>(maxe, mine, sum, sumsq,
                                                             G[blk], Bt[blk], O, invcnt,
                                                             outp, coff[blk], xn[blk]);
    xcur = xn[blk];
  }
}
